// SetAbstraction_80083960201798
// MI455X (gfx1250) — matrix operand order repaired and checked by reading, not compiled, not run on hardware
//
#include <hip/hip_runtime.h>

#define B_    16
#define N_    4096
#define S_    1024
#define K_    32
#define DFEAT 64
#define CIN   67      // 3 + 64
#define CPAD0 96      // pad 67 -> 3 * 32 for WMMA K-chunks
#define M_    (B_*S_*K_)   // 524288 rows

typedef __attribute__((ext_vector_type(16))) _Float16 v16h;
typedef __attribute__((ext_vector_type(8)))  float    v8f;

union Frag16 { v16h v; _Float16 h[16]; uint4 q[2]; };
union Frag8f { v8f  v; float    f[8];  };

// ---------------------------------------------------------------------------
// Kernel 1: farthest point sampling. One block (256 thr) per batch.
// ---------------------------------------------------------------------------
__global__ void fps_kernel(const float* __restrict__ xyz,
                           float* __restrict__ out_newxyz,
                           float* __restrict__ out_fpsidx) {
  __shared__ float px[N_], py[N_], pz[N_];
  __shared__ float rmax[256];
  __shared__ int   ridx[256];
  __shared__ int   s_far;
  const int b = blockIdx.x, tid = threadIdx.x;
  const float* base = xyz + (size_t)b * N_ * 3;
  for (int i = tid; i < N_; i += 256) {
    px[i] = base[i*3+0]; py[i] = base[i*3+1]; pz[i] = base[i*3+2];
  }
  float d[16];
#pragma unroll
  for (int j = 0; j < 16; ++j) d[j] = 1e10f;
  __syncthreads();

  int far = 0;
  if (tid == 0) {
    out_fpsidx[b*S_] = 0.0f;
    out_newxyz[(size_t)(b*S_)*3+0] = px[0];
    out_newxyz[(size_t)(b*S_)*3+1] = py[0];
    out_newxyz[(size_t)(b*S_)*3+2] = pz[0];
  }
  for (int t = 1; t < S_; ++t) {
    const float cx = px[far], cy = py[far], cz = pz[far];
    float best = -1.0f; int bidx = 0;
#pragma unroll
    for (int j = 0; j < 16; ++j) {
      const int i = tid + j*256;
      const float dx = px[i]-cx, dy = py[i]-cy, dz = pz[i]-cz;
      const float dd = dx*dx + dy*dy + dz*dz;
      d[j] = fminf(d[j], dd);
      if (d[j] > best) { best = d[j]; bidx = i; }   // strictly > => first index wins
    }
    rmax[tid] = best; ridx[tid] = bidx;
    __syncthreads();
    if (tid < 32) {                                  // wave0 reduces 256 candidates
      float v = -1.0f; int vi = 1 << 30;
      for (int e = 0; e < 8; ++e) {
        const int p = tid*8 + e;
        const float rv = rmax[p]; const int rvi = ridx[p];
        if (rv > v || (rv == v && rvi < vi)) { v = rv; vi = rvi; }
      }
      for (int off = 16; off; off >>= 1) {
        const float ov = __shfl_xor(v, off);
        const int   oi = __shfl_xor(vi, off);
        if (ov > v || (ov == v && oi < vi)) { v = ov; vi = oi; }
      }
      if (tid == 0) {
        s_far = vi;
        out_fpsidx[b*S_ + t] = (float)vi;
        const size_t o = ((size_t)b*S_ + t) * 3;
        out_newxyz[o+0] = px[vi]; out_newxyz[o+1] = py[vi]; out_newxyz[o+2] = pz[vi];
      }
    }
    __syncthreads();
    far = s_far;
  }
}

// ---------------------------------------------------------------------------
// Kernel 2: KNN (32 nearest of 4096) + gather + concat -> padded f16 X0.
// Block = 64 threads = 2 waves, one query per wave, 16KB LDS distances each.
// ---------------------------------------------------------------------------
__global__ void knn_gather_kernel(const float* __restrict__ xyz,
                                  const float* __restrict__ points,
                                  const float* __restrict__ newxyz,
                                  _Float16* __restrict__ X0) {
  __shared__ float dist[2][N_];
  const int w = threadIdx.x >> 5, lane = threadIdx.x & 31;
  const int qid = blockIdx.x * 2 + w;          // < 16384 always
  const int b = qid / S_;
  const float* pb = xyz + (size_t)b * N_ * 3;
  const float cx = newxyz[(size_t)qid*3+0];
  const float cy = newxyz[(size_t)qid*3+1];
  const float cz = newxyz[(size_t)qid*3+2];
  for (int i = lane; i < N_; i += 32) {
    const float dx = pb[i*3+0]-cx, dy = pb[i*3+1]-cy, dz = pb[i*3+2]-cz;
    dist[w][i] = dx*dx + dy*dy + dz*dz;
  }
  __syncthreads();
  const float* fb = points + (size_t)b * N_ * DFEAT;
  for (int k = 0; k < K_; ++k) {
    float bv = 3e38f; int bi = 1 << 30;
    for (int i = lane; i < N_; i += 32) {
      const float v = dist[w][i];
      if (v < bv || (v == bv && i < bi)) { bv = v; bi = i; }
    }
    for (int off = 16; off; off >>= 1) {         // butterfly: all lanes get winner
      const float ov = __shfl_xor(bv, off);
      const int   oi = __shfl_xor(bi, off);
      if (ov < bv || (ov == bv && oi < bi)) { bv = ov; bi = oi; }
    }
    const int j = bi;
    if (lane == 0) dist[w][j] = 3e38f;           // mask selected
    _Float16* xr = X0 + ((size_t)qid * K_ + k) * CPAD0;
    if (lane < 3)
      xr[lane] = (_Float16)(pb[j*3+lane] - (lane==0 ? cx : (lane==1 ? cy : cz)));
    xr[3 + lane]      = (_Float16)fb[(size_t)j*DFEAT + lane];
    xr[3 + 32 + lane] = (_Float16)fb[(size_t)j*DFEAT + 32 + lane];
    if (lane < CPAD0 - CIN) xr[CIN + lane] = (_Float16)0.0f;   // zero pad 67..95
    __syncthreads();
  }
}

// ---------------------------------------------------------------------------
// Kernel 3: WMMA GEMM  Y[M,O] = X[M,CPAD] * W^T  (bias dropped: BN absorbs it)
// plus f32 per-channel sum / sumsq accumulation for train-mode BN stats.
//
// N-partitioned: each wave owns NTW n-tiles, builds its B fragments ONCE into
// registers (loop-invariant, <=48 VGPRs -> no scratch spills), then streams
// 16-row A strips. A re-reads across NGROUPS waves hit L2 (X fits in 192MB).
// ---------------------------------------------------------------------------
template<int CPAD, int O, int NTW>
__global__ void __launch_bounds__(256)
gemm_wmma_kernel(const _Float16* __restrict__ X,
                 const float* __restrict__ Wf, int Cvalid,
                 _Float16* __restrict__ Y,
                 float* __restrict__ sum, float* __restrict__ sumsq) {
  constexpr int NK      = CPAD / 32;
  constexpr int NGROUPS = (O / 16) / NTW;
  __shared__ __align__(16) _Float16 Wlds[O * CPAD];
  const int tid = threadIdx.x;
  for (int i = tid; i < O * CPAD; i += 256) {    // stage weights (zero-padded) in LDS
    const int o = i / CPAD, c = i % CPAD;
    Wlds[i] = (c < Cvalid) ? (_Float16)Wf[o * Cvalid + c] : (_Float16)0.0f;
  }
  __syncthreads();

  const int wid = tid >> 5, lane = tid & 31;
  const int lr = lane & 15, lh = lane >> 4;
  const int wave_global = blockIdx.x * 8 + wid;
  const int group = wave_global % NGROUPS;       // which n-tile group this wave owns
  const int sidx  = wave_global / NGROUPS;
  const int sstride = (gridDim.x * 8) / NGROUPS;

  // Build this wave's B fragments once: B[N=lane%16][K = lh*16 + h]
  Frag16 bfr[NTW][NK];
#pragma unroll
  for (int nt = 0; nt < NTW; ++nt) {
    const int o = (group * NTW + nt) * 16 + lr;
#pragma unroll
    for (int kc = 0; kc < NK; ++kc) {
      const _Float16* wp = &Wlds[o * CPAD + kc * 32 + lh * 16];
      bfr[nt][kc].q[0] = *(const uint4*)wp;
      bfr[nt][kc].q[1] = *(const uint4*)(wp + 8);
    }
  }

  float accS[NTW], accQ[NTW];
#pragma unroll
  for (int nt = 0; nt < NTW; ++nt) { accS[nt] = 0.0f; accQ[nt] = 0.0f; }

  const int NSTRIP = M_ / 16;                    // 32768
  for (int strip = sidx; strip < NSTRIP; strip += sstride) {
    if (strip + sstride < NSTRIP)                // global_prefetch_b8 next A strip
      __builtin_prefetch(X + (size_t)((strip + sstride) * 16 + lr) * CPAD, 0, 1);

    Frag16 a[NK];                                // A: row = strip row, K contiguous: k = lh*16 + s
    const _Float16* xrow = X + (size_t)(strip * 16 + lr) * CPAD;
#pragma unroll
    for (int kc = 0; kc < NK; ++kc) {
      const _Float16* p = xrow + kc * 32 + lh * 16;
      a[kc].q[0] = *(const uint4*)p;             // halves 0..7  -> K +0..7
      a[kc].q[1] = *(const uint4*)(p + 8);       // halves 8..15 -> K +8..15
    }
#pragma unroll
    for (int nt = 0; nt < NTW; ++nt) {
      Frag8f acc;
#pragma unroll
      for (int r = 0; r < 8; ++r) acc.f[r] = 0.0f;
      const int o = (group * NTW + nt) * 16 + lr;   // D column for this lane
#pragma unroll
      for (int kc = 0; kc < NK; ++kc) {
        acc.v = __builtin_amdgcn_wmma_f32_16x16x32_f16(
            false, a[kc].v, false, bfr[nt][kc].v, (short)0, acc.v, false, false);
      }
      const int m0 = strip * 16 + lh * 8;
      float ls = 0.0f, lq = 0.0f;
#pragma unroll
      for (int r = 0; r < 8; ++r) {
        const float v = acc.f[r];
        Y[(size_t)(m0 + r) * O + o] = (_Float16)v;
        ls += v; lq += v * v;
      }
      accS[nt] += ls; accQ[nt] += lq;
    }
  }
  // lanes L and L+16 share the same channel: one xor-16 add, then atomics
#pragma unroll
  for (int nt = 0; nt < NTW; ++nt) {
    const float s = accS[nt] + __shfl_xor(accS[nt], 16);
    const float q = accQ[nt] + __shfl_xor(accQ[nt], 16);
    if (lh == 0) {
      const int ch = (group * NTW + nt) * 16 + lr;
      atomicAdd(&sum[ch], s);
      atomicAdd(&sumsq[ch], q);
    }
  }
}

// ---------------------------------------------------------------------------
// Stats finalize: scale = gamma * rsqrt(var+eps), shift = beta - mean*scale
// ---------------------------------------------------------------------------
__global__ void bn_stats_kernel(const float* __restrict__ sum, const float* __restrict__ sumsq,
                                const float* __restrict__ gamma, const float* __restrict__ beta,
                                float* __restrict__ scale, float* __restrict__ shift, int O) {
  const int o = threadIdx.x;
  if (o < O) {
    const float inv  = 1.0f / (float)M_;
    const float mean = sum[o] * inv;
    const float var  = sumsq[o] * inv - mean * mean;   // biased var, bias cancelled
    const float sc   = gamma[o] * rsqrtf(var + 1e-5f);
    scale[o] = sc;
    shift[o] = beta[o] - mean * sc;
  }
}

__global__ void bn_relu_kernel(const _Float16* __restrict__ Y,
                               const float* __restrict__ scale, const float* __restrict__ shift,
                               _Float16* __restrict__ Xn, int O, size_t total) {
  const size_t i = (size_t)blockIdx.x * blockDim.x + threadIdx.x;
  if (i < total) {
    const int o = (int)(i % (size_t)O);
    const float v = scale[o] * (float)Y[i] + shift[o];
    Xn[i] = (_Float16)fmaxf(v, 0.0f);
  }
}

// Last layer: BN + ReLU + max over K neighbors -> new_points (f32)
__global__ void bn_relu_maxpool_kernel(const _Float16* __restrict__ Y,
                                       const float* __restrict__ scale,
                                       const float* __restrict__ shift,
                                       float* __restrict__ out) {
  const int idx = blockIdx.x * blockDim.x + threadIdx.x;  // B*S*128 threads
  const int o = idx & 127, bs = idx >> 7;
  const float sc = scale[o], sh = shift[o];
  float m = 0.0f;                                         // relu outputs are >= 0
  for (int k = 0; k < K_; ++k) {
    const float v = sc * (float)Y[((size_t)bs * K_ + k) * 128 + o] + sh;
    m = fmaxf(m, fmaxf(v, 0.0f));
  }
  out[idx] = m;
}

__global__ void zero_kernel(float* p, int n) {
  const int i = blockIdx.x * 256 + threadIdx.x;
  if (i < n) p[i] = 0.0f;
}

// ---------------------------------------------------------------------------
extern "C" void kernel_launch(void* const* d_in, const int* in_sizes, int n_in,
                              void* d_out, int out_size, void* d_ws, size_t ws_size,
                              hipStream_t stream) {
  (void)in_sizes; (void)n_in; (void)out_size; (void)ws_size;
  const float* xyz    = (const float*)d_in[0];
  const float* points = (const float*)d_in[1];
  // params flattened: (W, b, gamma, beta) x 3 ; bias unused (absorbed by BN)
  const float* W1 = (const float*)d_in[2];
  const float* g1 = (const float*)d_in[4];  const float* be1 = (const float*)d_in[5];
  const float* W2 = (const float*)d_in[6];
  const float* g2 = (const float*)d_in[8];  const float* be2 = (const float*)d_in[9];
  const float* W3 = (const float*)d_in[10];
  const float* g3 = (const float*)d_in[12]; const float* be3 = (const float*)d_in[13];

  float* out        = (float*)d_out;
  float* out_newxyz = out;                                   // [16,1024,3]
  float* out_newpts = out + (size_t)B_*S_*3;                 // [16,1024,128]
  float* out_fps    = out + (size_t)B_*S_*3 + (size_t)B_*S_*128;  // [16,1024]

  char* wsb = (char*)d_ws;
  float* sum   = (float*)wsb;        // 128 f32
  float* sumsq = sum + 128;
  float* scale = sum + 256;
  float* shift = sum + 384;
  _Float16* X0 = (_Float16*)(wsb + 4096);                               // M x 96
  _Float16* X1 = (_Float16*)(wsb + 4096 + (size_t)M_*CPAD0*2);          // M x 64
  _Float16* Yb = (_Float16*)(wsb + 4096 + (size_t)M_*CPAD0*2
                                         + (size_t)M_*64*2);            // M x 128 max

  // 1) FPS -> new_xyz + fps_idx
  fps_kernel<<<B_, 256, 0, stream>>>(xyz, out_newxyz, out_fps);
  // 2) KNN + gather + concat -> X0 (f16, padded to 96 ch)
  knn_gather_kernel<<<B_*S_/2, 64, 0, stream>>>(xyz, points, out_newxyz, X0);

  // Layer 1: 67(->96) -> 64
  zero_kernel<<<1, 256, 0, stream>>>(sum, 256);
  gemm_wmma_kernel<CPAD0, 64, 2><<<512, 256, 0, stream>>>(X0, W1, CIN, Yb, sum, sumsq);
  bn_stats_kernel<<<1, 128, 0, stream>>>(sum, sumsq, g1, be1, scale, shift, 64);
  bn_relu_kernel<<<(int)(((size_t)M_*64 + 255)/256), 256, 0, stream>>>(
      Yb, scale, shift, X1, 64, (size_t)M_*64);

  // Layer 2: 64 -> 64
  zero_kernel<<<1, 256, 0, stream>>>(sum, 256);
  gemm_wmma_kernel<64, 64, 2><<<512, 256, 0, stream>>>(X1, W2, 64, Yb, sum, sumsq);
  bn_stats_kernel<<<1, 128, 0, stream>>>(sum, sumsq, g2, be2, scale, shift, 64);
  bn_relu_kernel<<<(int)(((size_t)M_*64 + 255)/256), 256, 0, stream>>>(
      Yb, scale, shift, X1, 64, (size_t)M_*64);   // X1 overwritten with layer-2 output

  // Layer 3: 64 -> 128
  zero_kernel<<<1, 256, 0, stream>>>(sum, 256);
  gemm_wmma_kernel<64, 128, 2><<<512, 256, 0, stream>>>(X1, W3, 64, Yb, sum, sumsq);
  bn_stats_kernel<<<1, 128, 0, stream>>>(sum, sumsq, g3, be3, scale, shift, 128);
  // BN + ReLU + max over K -> new_points
  bn_relu_maxpool_kernel<<<B_*S_*128/256, 256, 0, stream>>>(Yb, scale, shift, out_newpts);
}
